// GroupedQueryAttention_9526237462741
// MI455X (gfx1250) — matrix operand order repaired and checked by reading, not compiled, not run on hardware
//
#include <hip/hip_runtime.h>

// ---------- CDNA5 WMMA types ----------
typedef __attribute__((ext_vector_type(16))) __bf16 v16bf;
typedef __attribute__((ext_vector_type(8)))  float  v8f;
typedef __attribute__((ext_vector_type(4)))  unsigned int u32x4;

union FragBF {
  u32x4 u[2];   // two 16B chunks (8 bf16 each)
  v16bf v;      // 16 bf16 operand for v_wmma_f32_16x16x32_bf16
};

__device__ __forceinline__ unsigned short f2bf(float f) {
  union { float f; unsigned int u; } c; c.f = f;
  unsigned int u = c.u;
  return (unsigned short)((u + 0x7FFFu + ((u >> 16) & 1u)) >> 16);  // RNE
}

// ---------- fp32 -> bf16 cast ----------
__global__ void cast_f32_bf16(const float* __restrict__ src,
                              unsigned short* __restrict__ dst, int n) {
  int i = (blockIdx.x * blockDim.x + threadIdx.x) * 4;
  if (i + 3 < n) {
    float4 v = *(const float4*)(src + i);
    dst[i + 0] = f2bf(v.x); dst[i + 1] = f2bf(v.y);
    dst[i + 2] = f2bf(v.z); dst[i + 3] = f2bf(v.w);
  } else {
    for (int k = i; k < n; ++k) dst[k] = f2bf(src[k]);
  }
}

// ---------- tiled bf16 WMMA GEMM:  C[M,N] = A[M,K] * B[K,N] + bias, *scale ----------
// block = 256 threads (8 waves), tile 128x128, k-step 32.
// Double-buffered LDS, register-staged global loads, ONE barrier per k-step.
// wave (wm,wn) in 4x2 grid computes 32x64 = 2x4 tiles of 16x16.
template <bool OUT_F32>
__global__ __launch_bounds__(256)
void gemm_bf16_wmma(const unsigned short* __restrict__ A,
                    const unsigned short* __restrict__ Bm,
                    const float* __restrict__ bias,
                    void* __restrict__ Cp,
                    int M, int N, int K, float scaleOut) {
  __shared__ __align__(16) unsigned short As[2][128][48];   // 32 cols + pad (96B stride)
  __shared__ __align__(16) unsigned short Bs[2][32][144];   // 128 cols + pad (288B stride)

  const int tid  = threadIdx.x;
  const int wave = tid >> 5;
  const int lane = tid & 31;
  const int half = lane >> 4;
  const int l16  = lane & 15;

  const int m0 = blockIdx.y * 128;
  const int n0 = blockIdx.x * 128;
  const int m0w = (wave >> 1) * 32;   // 0..96
  const int n0w = (wave & 1) * 64;    // 0/64

  v8f acc[2][4];
  for (int i = 0; i < 2; ++i)
    for (int j = 0; j < 4; ++j) acc[i][j] = {};

  const int ar = tid >> 1, ac = (tid & 1) * 16;   // A: 128 rows x 32 cols
  const int br = tid >> 3, bc = (tid & 7) * 16;   // B: 32 rows x 128 cols

  const unsigned short* gA = A  + (size_t)(m0 + ar) * K + ac;   // + k0
  const unsigned short* gB0 = Bm + (size_t)br * N + n0 + bc;    // + k0*N

  u32x4 ra[2], rb[2];
  // ---- preload tile 0 into LDS buffer 0 ----
  ra[0] = *(const u32x4*)gA;            ra[1] = *(const u32x4*)(gA + 8);
  rb[0] = *(const u32x4*)gB0;           rb[1] = *(const u32x4*)(gB0 + 8);
  *(u32x4*)&As[0][ar][ac]     = ra[0];  *(u32x4*)&As[0][ar][ac + 8] = ra[1];
  *(u32x4*)&Bs[0][br][bc]     = rb[0];  *(u32x4*)&Bs[0][br][bc + 8] = rb[1];
  __syncthreads();

  const int nIter = K >> 5;
  for (int it = 0; it < nIter; ++it) {
    const int cur = it & 1;
    const bool hasNext = (it + 1) < nIter;

    // stage next tile's global loads (overlaps with WMMA below)
    if (hasNext) {
      const int kn = (it + 1) << 5;
      ra[0] = *(const u32x4*)(gA + kn);
      ra[1] = *(const u32x4*)(gA + kn + 8);
      rb[0] = *(const u32x4*)(gB0 + (size_t)kn * N);
      rb[1] = *(const u32x4*)(gB0 + (size_t)kn * N + 8);
      if (it + 2 < nIter) {   // warm L2 two steps ahead (global_prefetch_b8)
        const int kp = (it + 2) << 5;
        __builtin_prefetch(gA + kp, 0, 0);
        __builtin_prefetch(gB0 + (size_t)kp * N, 0, 0);
      }
    }

    // ---- compute on current buffer ----
    FragBF afrag[2];
#pragma unroll
    for (int mi = 0; mi < 2; ++mi) {
      const unsigned short* p = &As[cur][m0w + mi * 16 + l16][0];
      afrag[mi].u[0] = *(const u32x4*)(p + half * 8);
      afrag[mi].u[1] = *(const u32x4*)(p + 16 + half * 8);
    }
    FragBF bfrag[4];
#pragma unroll
    for (int ni = 0; ni < 4; ++ni) {
      __attribute__((aligned(16))) unsigned short t[16];
#pragma unroll
      for (int s = 0; s < 16; ++s)
        t[s] = Bs[cur][(s < 8) ? (half * 8 + s) : (16 + half * 8 + s - 8)][n0w + ni * 16 + l16];
      bfrag[ni].u[0] = *(const u32x4*)&t[0];
      bfrag[ni].u[1] = *(const u32x4*)&t[8];
    }
#pragma unroll
    for (int mi = 0; mi < 2; ++mi)
#pragma unroll
      for (int ni = 0; ni < 4; ++ni)
        acc[mi][ni] = __builtin_amdgcn_wmma_f32_16x16x32_bf16(
            false, afrag[mi].v, false, bfrag[ni].v, (short)0, acc[mi][ni],
            false, false);

    // ---- publish next buffer; single barrier per k-step ----
    if (hasNext) {
      const int nxt = cur ^ 1;
      *(u32x4*)&As[nxt][ar][ac]     = ra[0];
      *(u32x4*)&As[nxt][ar][ac + 8] = ra[1];
      *(u32x4*)&Bs[nxt][br][bc]     = rb[0];
      *(u32x4*)&Bs[nxt][br][bc + 8] = rb[1];
      __syncthreads();
    }
  }

  // epilogue: C layout = col (lane&15), rows r + 8*half
#pragma unroll
  for (int mi = 0; mi < 2; ++mi) {
#pragma unroll
    for (int ni = 0; ni < 4; ++ni) {
      const int gn = n0 + n0w + ni * 16 + l16;
      const float bv = bias ? bias[gn] : 0.f;
#pragma unroll
      for (int r = 0; r < 8; ++r) {
        const int gm = m0 + m0w + mi * 16 + r + 8 * half;
        const float v = (acc[mi][ni][r] + bv) * scaleOut;
        if (OUT_F32) ((float*)Cp)[(size_t)gm * N + gn] = v;
        else ((unsigned short*)Cp)[(size_t)gm * N + gn] = f2bf(v);
      }
    }
  }
}

// ---------- causal flash attention (GQA), bf16 WMMA, fp32 accum ----------
// block = 256 threads (8 waves) handles (b, head h, 128 query rows).
// Each wave owns 16 query rows; KV streamed in 64-row tiles, double-buffered LDS.
__global__ __launch_bounds__(256)
void fmha_gqa(const unsigned short* __restrict__ Qb,
              const unsigned short* __restrict__ Kb,
              const unsigned short* __restrict__ Vb,
              unsigned short* __restrict__ Ctx) {
  constexpr int S = 2048, DK = 64, QSTR = 2048, KSTR = 512;

  __shared__ __align__(16) unsigned short Kt[2][64][72];    // transposed: Kt[.][d][kv]
  __shared__ __align__(16) unsigned short Vs[2][64][72];    // Vs[.][kv][d]
  __shared__ __align__(16) unsigned short Ps[8][16][72];    // per-wave P staging

  const int tid  = threadIdx.x;
  const int wave = tid >> 5;
  const int lane = tid & 31;
  const int half = lane >> 4;
  const int l16  = lane & 15;

  const int qbase = blockIdx.x * 128;
  const int h = blockIdx.y;
  const int b = blockIdx.z;
  const int g = h >> 2;              // REP = 4 -> kv group

  const int m0w = wave * 16;
  const int qlrow = qbase + m0w + l16;   // A-fragment row for this lane

  // Q fragments (two 16x32 k-slices of the 16x64 Q tile), loaded straight from global
  FragBF qa[2];
  {
    const unsigned short* qp = Qb + (size_t)(b * S + qlrow) * QSTR + h * DK;
#pragma unroll
    for (int kk = 0; kk < 2; ++kk) {
      qa[kk].u[0] = *(const u32x4*)(qp + kk * 32 + half * 8);
      qa[kk].u[1] = *(const u32x4*)(qp + kk * 32 + 16 + half * 8);
    }
  }

  v8f o[4];
  for (int j = 0; j < 4; ++j) o[j] = {};
  float m_i[8], l_i[8];
#pragma unroll
  for (int r = 0; r < 8; ++r) { m_i[r] = -1e30f; l_i[r] = 0.f; }

  // per-thread KV tile load coords
  const int kj = tid & 63, kd0 = (tid >> 6) * 16;   // K: row kv+kj, dims kd0..+15
  const int vj = tid >> 2, vd0 = (tid & 3) * 16;    // V: row kv+vj, dims vd0..+15
  const unsigned short* gK = Kb + (size_t)(b * S + kj) * KSTR + g * DK + kd0;
  const unsigned short* gV = Vb + (size_t)(b * S + vj) * KSTR + g * DK + vd0;

  u32x4 kr[2], vr[2];
  // ---- preload KV tile 0 into buffer 0 ----
  kr[0] = *(const u32x4*)gK;  kr[1] = *(const u32x4*)(gK + 8);
  vr[0] = *(const u32x4*)gV;  vr[1] = *(const u32x4*)(gV + 8);
  {
    unsigned short tmp[16];
    *(u32x4*)&tmp[0] = kr[0]; *(u32x4*)&tmp[8] = kr[1];
#pragma unroll
    for (int e = 0; e < 16; ++e) Kt[0][kd0 + e][kj] = tmp[e];   // transpose on store
    *(u32x4*)&Vs[0][vj][vd0]     = vr[0];
    *(u32x4*)&Vs[0][vj][vd0 + 8] = vr[1];
  }
  __syncthreads();

  const int nIter = (qbase + 128) >> 6;   // causal: never need KV past query block
  for (int it = 0; it < nIter; ++it) {
    const int kv0 = it << 6;
    const int cur = it & 1;
    const bool hasNext = (it + 1) < nIter;

    // stage next KV tile (global loads overlap with WMMA/softmax below)
    if (hasNext) {
      const size_t off = (size_t)(it + 1) * 64 * KSTR;
      kr[0] = *(const u32x4*)(gK + off);  kr[1] = *(const u32x4*)(gK + off + 8);
      vr[0] = *(const u32x4*)(gV + off);  vr[1] = *(const u32x4*)(gV + off + 8);
      if (it + 2 < nIter) {   // global_prefetch_b8 two tiles ahead
        const size_t off2 = (size_t)(it + 2) * 64 * KSTR;
        __builtin_prefetch(gK + off2, 0, 0);
        __builtin_prefetch(gV + off2, 0, 0);
      }
    }

    if (kv0 < qbase + m0w + 16) {   // skip blocks entirely in the causal future
      // ---- S = Q * K^T (16x64, 8 WMMAs) ----
      v8f s4[4];
      for (int j = 0; j < 4; ++j) s4[j] = {};
#pragma unroll
      for (int kk = 0; kk < 2; ++kk)
#pragma unroll
        for (int j = 0; j < 4; ++j) {
          FragBF kf;
          __attribute__((aligned(16))) unsigned short t[16];
#pragma unroll
          for (int s = 0; s < 16; ++s)
            t[s] = Kt[cur][kk * 32 + ((s < 8) ? (half * 8 + s) : (16 + half * 8 + s - 8))][j * 16 + l16];
          kf.u[0] = *(const u32x4*)&t[0];
          kf.u[1] = *(const u32x4*)&t[8];
          s4[j] = __builtin_amdgcn_wmma_f32_16x16x32_bf16(
              false, qa[kk].v, false, kf.v, (short)0, s4[j], false, false);
        }

      // ---- causal mask + online softmax (row = r + 8*half, col = lane&15) ----
      float newm[8];
#pragma unroll
      for (int r = 0; r < 8; ++r) {
        const int qr = qbase + m0w + r + 8 * half;
        float mx = m_i[r];
#pragma unroll
        for (int j = 0; j < 4; ++j) {
          const int col = kv0 + j * 16 + l16;
          float v = (col <= qr) ? s4[j][r] : -1e30f;
          s4[j][r] = v;
          mx = fmaxf(mx, v);
        }
#pragma unroll
        for (int x = 1; x < 16; x <<= 1) mx = fmaxf(mx, __shfl_xor(mx, x, 32));
        newm[r] = mx;
      }
#pragma unroll
      for (int r = 0; r < 8; ++r) {
        float rs = 0.f;
#pragma unroll
        for (int j = 0; j < 4; ++j) {
          const float p = __expf(s4[j][r] - newm[r]);
          s4[j][r] = p;
          rs += p;
        }
#pragma unroll
        for (int x = 1; x < 16; x <<= 1) rs += __shfl_xor(rs, x, 32);
        const float sc = __expf(m_i[r] - newm[r]);
        l_i[r] = l_i[r] * sc + rs;
        m_i[r] = newm[r];
#pragma unroll
        for (int j = 0; j < 4; ++j) o[j][r] = o[j][r] * sc;
      }

      // ---- stage P (C layout) to LDS row-major, re-read as A fragments ----
      // per-wave private region: no cross-wave barrier needed (dscnt wait only)
#pragma unroll
      for (int j = 0; j < 4; ++j)
#pragma unroll
        for (int r = 0; r < 8; ++r)
          Ps[wave][r + 8 * half][j * 16 + l16] = f2bf(s4[j][r]);

      FragBF pa[2];
#pragma unroll
      for (int kk = 0; kk < 2; ++kk) {
        const unsigned short* p = &Ps[wave][l16][kk * 32];
        pa[kk].u[0] = *(const u32x4*)(p + half * 8);
        pa[kk].u[1] = *(const u32x4*)(p + 16 + half * 8);
      }

      // ---- O += P * V (8 WMMAs) ----
#pragma unroll
      for (int kk = 0; kk < 2; ++kk)
#pragma unroll
        for (int j = 0; j < 4; ++j) {
          FragBF vf;
          __attribute__((aligned(16))) unsigned short t[16];
#pragma unroll
          for (int s = 0; s < 16; ++s)
            t[s] = Vs[cur][kk * 32 + ((s < 8) ? (half * 8 + s) : (16 + half * 8 + s - 8))][j * 16 + l16];
          vf.u[0] = *(const u32x4*)&t[0];
          vf.u[1] = *(const u32x4*)&t[8];
          o[j] = __builtin_amdgcn_wmma_f32_16x16x32_bf16(
              false, pa[kk].v, false, vf.v, (short)0, o[j], false, false);
        }
    }

    // ---- publish next KV buffer; single barrier per tile ----
    if (hasNext) {
      const int nxt = cur ^ 1;
      unsigned short tmp[16];
      *(u32x4*)&tmp[0] = kr[0]; *(u32x4*)&tmp[8] = kr[1];
#pragma unroll
      for (int e = 0; e < 16; ++e) Kt[nxt][kd0 + e][kj] = tmp[e];
      *(u32x4*)&Vs[nxt][vj][vd0]     = vr[0];
      *(u32x4*)&Vs[nxt][vj][vd0 + 8] = vr[1];
      __syncthreads();
    }
  }

  // ---- normalize + store context (bf16) ----
#pragma unroll
  for (int j = 0; j < 4; ++j)
#pragma unroll
    for (int r = 0; r < 8; ++r) {
      const int qr = qbase + m0w + r + 8 * half;
      const float v = o[j][r] / l_i[r];
      Ctx[(size_t)(b * S + qr) * QSTR + h * DK + j * 16 + l16] = f2bf(v);
    }
}

// ---------- host orchestration ----------
extern "C" void kernel_launch(void* const* d_in, const int* in_sizes, int n_in,
                              void* d_out, int out_size, void* d_ws, size_t ws_size,
                              hipStream_t stream) {
  (void)in_sizes; (void)n_in; (void)out_size; (void)ws_size;

  const float* x  = (const float*)d_in[0];
  const float* Wq = (const float*)d_in[1];
  const float* bq = (const float*)d_in[2];
  const float* Wk = (const float*)d_in[3];
  const float* bk = (const float*)d_in[4];
  const float* Wv = (const float*)d_in[5];
  const float* bv = (const float*)d_in[6];
  const float* Wo = (const float*)d_in[7];
  const float* bo = (const float*)d_in[8];
  float* out = (float*)d_out;

  constexpr int Bd = 2, S = 2048, D = 2048, H = 32, KV = 8, DK = 64;
  constexpr int M = Bd * S;                    // 4096
  constexpr int NKV = KV * DK;                 // 512
  constexpr size_t nX  = (size_t)M * D;        // 8388608
  constexpr size_t nWq = (size_t)D * D;        // 4194304
  constexpr size_t nWk = (size_t)D * NKV;      // 1048576
  constexpr size_t nKV = (size_t)M * NKV;      // 2097152

  // workspace layout (bf16 buffers), ~80 MB total
  char* ws = (char*)d_ws;
  unsigned short* xb  = (unsigned short*)ws; ws += nX  * 2;
  unsigned short* Wqb = (unsigned short*)ws; ws += nWq * 2;
  unsigned short* Wkb = (unsigned short*)ws; ws += nWk * 2;
  unsigned short* Wvb = (unsigned short*)ws; ws += nWk * 2;
  unsigned short* Wob = (unsigned short*)ws; ws += nWq * 2;
  unsigned short* Qb  = (unsigned short*)ws; ws += nX  * 2;
  unsigned short* Kb  = (unsigned short*)ws; ws += nKV * 2;
  unsigned short* Vb  = (unsigned short*)ws; ws += nKV * 2;
  unsigned short* Ctx = (unsigned short*)ws; ws += nX  * 2;

  auto cast = [&](const float* s, unsigned short* d, size_t n) {
    int blocks = (int)((n / 4 + 255) / 256);
    cast_f32_bf16<<<blocks, 256, 0, stream>>>(s, d, (int)n);
  };
  cast(x,  xb,  nX);
  cast(Wq, Wqb, nWq);
  cast(Wk, Wkb, nWk);
  cast(Wv, Wvb, nWk);
  cast(Wo, Wob, nWq);

  const dim3 blk(256);
  // Q = (x Wq + bq) * 1/sqrt(DK)  (scale folded into projection)
  gemm_bf16_wmma<false><<<dim3(D / 128, M / 128), blk, 0, stream>>>(
      xb, Wqb, bq, Qb, M, D, D, 0.125f);
  gemm_bf16_wmma<false><<<dim3(NKV / 128, M / 128), blk, 0, stream>>>(
      xb, Wkb, bk, Kb, M, NKV, D, 1.0f);
  gemm_bf16_wmma<false><<<dim3(NKV / 128, M / 128), blk, 0, stream>>>(
      xb, Wvb, bv, Vb, M, NKV, D, 1.0f);

  fmha_gqa<<<dim3(S / 128, H, Bd), blk, 0, stream>>>(Qb, Kb, Vb, Ctx);

  gemm_bf16_wmma<true><<<dim3(D / 128, M / 128), blk, 0, stream>>>(
      Ctx, Wob, bo, out, M, D, D, 1.0f);
}
